// InvariantHeadviaTP_6665789243829
// MI455X (gfx1250) — matrix operand order repaired and checked by reading, not compiled, not run on hardware
//
#include <hip/hip_runtime.h>
#include <hip/hip_bf16.h>

typedef __attribute__((ext_vector_type(16))) _Float16 v16h;
typedef __attribute__((ext_vector_type(2)))  _Float16 h2;
typedef __attribute__((ext_vector_type(8)))  _Float16 v8h;
typedef __attribute__((ext_vector_type(8)))  float    v8f;

#define NNODES 16384
#define AEMB   16
#define MUL0_  128
#define DIN_   480
#define OUT0_  224
#define KTOT   2048   // MUL0*AEMB contraction length
#define KSTEPS 64     // KTOT / 32 (K=32 per f16 WMMA)
#define MROWS  64     // rows per block (4 sub-tiles of 16) -> 4x B reuse
#define INV2048 0.02209708691207961f  // 1/sqrt(128*16), folded into weights

// ---------------------------------------------------------------------------
// Prep: repack live weight slabs to f16 (pre-scaled by 1/sqrt(2048)),
// column-major per output column so a lane's WMMA B-fragment is two
// 16B v8h loads, at k offsets 8h and 16+8h.
// ---------------------------------------------------------------------------
__global__ __launch_bounds__(256) void prep_weights(
    const float* __restrict__ W1,   // (128,16,224)
    const float* __restrict__ W2,   // (128,16,16)
    _Float16* __restrict__ W1h,     // [128][2048]
    _Float16* __restrict__ W2h)     // [16][2048]
{
  int idx = blockIdx.x * 256 + threadIdx.x;
  const int tot1 = MUL0_ * KTOT;                 // 262144
  if (idx < tot1) {
    int c = idx / KTOT, k = idx % KTOT;          // k = i*16+a
    W1h[idx] = (_Float16)(W1[(size_t)k * OUT0_ + c] * INV2048);
  }
  int idx2 = idx - tot1;
  if (idx2 >= 0 && idx2 < AEMB * KTOT) {
    int c = idx2 / KTOT, k = idx2 % KTOT;
    W2h[idx2] = (_Float16)(W2[(size_t)k * AEMB + c] * INV2048);
  }
}

// 16x32 f16 A-fragment for one K=32 step (8 v_pk_mul_f16):
//   lane<16 : row m, elem j<8 -> x_i0*e[j], elem j>=8 -> x_i1*e[j-8]
//   lane>=16: same with e[] = emb[row][8..15]
__device__ __forceinline__ v16h mk_af(h2 xa, h2 xb, h2 e01, h2 e23, h2 e45, h2 e67) {
  h2 p0 = xa * e01, p1 = xa * e23, p2 = xa * e45, p3 = xa * e67;
  h2 p4 = xb * e01, p5 = xb * e23, p6 = xb * e45, p7 = xb * e67;
  v16h af;
  af[0]  = p0[0]; af[1]  = p0[1]; af[2]  = p1[0]; af[3]  = p1[1];
  af[4]  = p2[0]; af[5]  = p2[1]; af[6]  = p3[0]; af[7]  = p3[1];
  af[8]  = p4[0]; af[9]  = p4[1]; af[10] = p5[0]; af[11] = p5[1];
  af[12] = p6[0]; af[13] = p6[1]; af[14] = p7[0]; af[15] = p7[1];
  return af;
}

// ---------------------------------------------------------------------------
// Fused main kernel: one block = 64 nodes (4 sub-tiles of 16 rows).
// Each wave owns 16 output columns and 4 accumulators; every B-fragment
// load feeds 4 WMMAs. All 4 A-fragments are built before the 4 WMMAs issue
// so no fragment register is rewritten until >=3 WMMAs later (kills the
// WMMA->VALU WAR-hazard v_nops).
// ---------------------------------------------------------------------------
__global__ __launch_bounds__(256) void fused_head(
    const float* __restrict__ node_vec,   // (N,480), only cols 0..127 live
    const float* __restrict__ emb,        // (N,16)
    const _Float16* __restrict__ W1h,
    const _Float16* __restrict__ W2h,
    const float* __restrict__ b1,
    const float* __restrict__ b2,
    const float* __restrict__ W3,         // (16,16)
    const float* __restrict__ b3,
    const float* __restrict__ W4,         // (16,1)
    const float* __restrict__ b4,
    float* __restrict__ out)              // (N)
{
  __shared__ h2    x0h[MROWS][130];  // half2(x,x) pre-splatted; +2 pad -> no conflicts
  __shared__ float embs[MROWS][16];
  __shared__ h2    sch[MROWS][130];  // half2(silu(s), silu(s))
  __shared__ float mids[MROWS][16];

  const int tid  = threadIdx.x;
  const int lane = tid & 31;
  const int wave = tid >> 5;
  const int n0   = blockIdx.x * MROWS;

  // ---- stage x0 tile (coalesced, convert+splat to half2) + emb tile ----
  for (int idx = tid; idx < MROWS * 128; idx += 256) {
    int r = idx >> 7, c = idx & 127;
    float v = node_vec[(size_t)(n0 + r) * DIN_ + c];
    _Float16 hv = (_Float16)v;
    h2 p; p[0] = hv; p[1] = hv;
    x0h[r][c] = p;
  }
  for (int idx = tid; idx < MROWS * 16; idx += 256) {
    int r = idx >> 4, c = idx & 15;
    embs[r][c] = emb[(size_t)(n0 + r) * AEMB + c];
  }
  __syncthreads();

  const int  m  = lane & 15;
  const bool hi = lane >= 16;

  // packed f16 emb pairs per sub-tile (constant across K loop)
  h2 e01[4], e23[4], e45[4], e67[4];
  #pragma unroll
  for (int t = 0; t < 4; ++t) {
    const float* er = &embs[16 * t + m][hi ? 8 : 0];
    e01[t][0] = (_Float16)er[0]; e01[t][1] = (_Float16)er[1];
    e23[t][0] = (_Float16)er[2]; e23[t][1] = (_Float16)er[3];
    e45[t][0] = (_Float16)er[4]; e45[t][1] = (_Float16)er[5];
    e67[t][0] = (_Float16)er[6]; e67[t][1] = (_Float16)er[7];
  }

  // ---- stage 1: s tiles, columns [16*wave, 16*wave+16), 4 row groups ----
  v8f acc[4] = {v8f{}, v8f{}, v8f{}, v8f{}};
  const int c0 = wave * 16;
  {
    const _Float16* bptr = W1h + (size_t)(c0 + m) * KTOT + (hi ? 8 : 0);
    v16h bf = __builtin_shufflevector(*(const v8h*)bptr, *(const v8h*)(bptr + 16), 0,1,2,3,4,5,6,7,8,9,10,11,12,13,14,15);                 // prologue B load
    h2 xa[4], xb[4];
    #pragma unroll
    for (int t = 0; t < 4; ++t) { xa[t] = x0h[16 * t + m][0]; xb[t] = x0h[16 * t + m][1]; }

    #pragma unroll 3
    for (int it = 0; it < KSTEPS - 1; ++it) {
      v16h bn = __builtin_shufflevector(*(const v8h*)(bptr + (it + 1) * 32), *(const v8h*)(bptr + (it + 1) * 32 + 16), 0,1,2,3,4,5,6,7,8,9,10,11,12,13,14,15);   // prefetch next B
      h2 xan[4], xbn[4];
      #pragma unroll
      for (int t = 0; t < 4; ++t) {
        xan[t] = x0h[16 * t + m][2 * it + 2];           // prefetch next x pairs
        xbn[t] = x0h[16 * t + m][2 * it + 3];
      }
      // build ALL fragments first (distinct regs), then issue 4 WMMAs
      v16h af[4];
      #pragma unroll
      for (int t = 0; t < 4; ++t)
        af[t] = mk_af(xa[t], xb[t], e01[t], e23[t], e45[t], e67[t]);
      #pragma unroll
      for (int t = 0; t < 4; ++t)
        acc[t] = __builtin_amdgcn_wmma_f32_16x16x32_f16(
            false, af[t], false, bf, (short)0, acc[t], false, false);
      bf = bn;
      #pragma unroll
      for (int t = 0; t < 4; ++t) { xa[t] = xan[t]; xb[t] = xbn[t]; }
    }
    v16h af[4];                                          // peeled last step
    #pragma unroll
    for (int t = 0; t < 4; ++t)
      af[t] = mk_af(xa[t], xb[t], e01[t], e23[t], e45[t], e67[t]);
    #pragma unroll
    for (int t = 0; t < 4; ++t)
      acc[t] = __builtin_amdgcn_wmma_f32_16x16x32_f16(
          false, af[t], false, bf, (short)0, acc[t], false, false);
  }

  // bias + silu (scale folded into W1h); C layout: row r+8*hi, col c0+m
  {
    float bb = b1[c0 + m];
    #pragma unroll
    for (int t = 0; t < 4; ++t) {
      #pragma unroll
      for (int r = 0; r < 8; ++r) {
        float s  = acc[t][r] + bb;
        float sl = s / (1.0f + __expf(-s));
        _Float16 hv = (_Float16)sl;
        h2 p; p[0] = hv; p[1] = hv;
        sch[16 * t + r + (hi ? 8 : 0)][c0 + m] = p;
      }
    }
  }
  __syncthreads();

  // ---- stage 2: mid = (silu(s) (x) emb) . W2 + b2; wave w owns sub-tile w ----
  if (wave < 4) {
    const int t = wave;
    v8f acc2 = {};
    const _Float16* bptr2 = W2h + (size_t)m * KTOT + (hi ? 8 : 0);
    v16h bf = __builtin_shufflevector(*(const v8h*)bptr2, *(const v8h*)(bptr2 + 16), 0,1,2,3,4,5,6,7,8,9,10,11,12,13,14,15);
    h2 xa = sch[16 * t + m][0], xb = sch[16 * t + m][1];
    #pragma unroll 7
    for (int it = 0; it < KSTEPS - 1; ++it) {
      v16h bn  = __builtin_shufflevector(*(const v8h*)(bptr2 + (it + 1) * 32), *(const v8h*)(bptr2 + (it + 1) * 32 + 16), 0,1,2,3,4,5,6,7,8,9,10,11,12,13,14,15);
      h2   xan = sch[16 * t + m][2 * it + 2];
      h2   xbn = sch[16 * t + m][2 * it + 3];
      v16h af = mk_af(xa, xb, e01[t], e23[t], e45[t], e67[t]);
      acc2 = __builtin_amdgcn_wmma_f32_16x16x32_f16(
          false, af, false, bf, (short)0, acc2, false, false);
      bf = bn; xa = xan; xb = xbn;
    }
    v16h af = mk_af(xa, xb, e01[t], e23[t], e45[t], e67[t]);
    acc2 = __builtin_amdgcn_wmma_f32_16x16x32_f16(
        false, af, false, bf, (short)0, acc2, false, false);

    float bb2 = b2[m];
    #pragma unroll
    for (int r = 0; r < 8; ++r)
      mids[16 * t + r + (hi ? 8 : 0)][m] = acc2[r] + bb2;
  }
  __syncthreads();

  // ---- stage 3: h = silu(mid @ W3 / 4 + b3); out = h @ W4 / 4 + b4 ----
  if (tid < MROWS) {
    float mrow[16];
    #pragma unroll
    for (int k = 0; k < 16; ++k) mrow[k] = mids[tid][k];
    float o = 0.0f;
    #pragma unroll
    for (int c = 0; c < 16; ++c) {
      float t3 = 0.0f;
      #pragma unroll
      for (int k = 0; k < 16; ++k) t3 += mrow[k] * W3[k * 16 + c];
      t3 = t3 * 0.25f + b3[c];                  // 0.25 = 1/sqrt(16)
      float hs = t3 / (1.0f + __expf(-t3));
      o += hs * W4[c];
    }
    out[n0 + tid] = o * 0.25f + b4[0];
  }
}

extern "C" void kernel_launch(void* const* d_in, const int* in_sizes, int n_in,
                              void* d_out, int out_size, void* d_ws, size_t ws_size,
                              hipStream_t stream) {
  const float* node_vec = (const float*)d_in[0];
  const float* node_emb = (const float*)d_in[1];
  const float* W1_l0    = (const float*)d_in[2];
  // d_in[3], d_in[4] (W1_l1, W1_l2) feed only dead outputs: skipped
  const float* b1       = (const float*)d_in[5];
  const float* W2       = (const float*)d_in[6];
  const float* b2       = (const float*)d_in[7];
  const float* W3       = (const float*)d_in[8];
  const float* b3       = (const float*)d_in[9];
  const float* W4       = (const float*)d_in[10];
  const float* b4       = (const float*)d_in[11];
  float* out = (float*)d_out;

  _Float16* W1h = (_Float16*)d_ws;                       // 128*2048 halves
  _Float16* W2h = W1h + (size_t)MUL0_ * KTOT;            // + 16*2048 halves

  const int prep_elems = MUL0_ * KTOT + AEMB * KTOT;     // 294912
  prep_weights<<<(prep_elems + 255) / 256, 256, 0, stream>>>(W1_l0, W2, W1h, W2h);
  fused_head<<<NNODES / MROWS, 256, 0, stream>>>(node_vec, node_emb, W1h, W2h,
                                                 b1, b2, W3, b3, W4, b4, out);
}
